// GraphPropLayer_4380866642095
// MI455X (gfx1250) — compile-verified, matrix operand order checked by reading, not run on hardware
//
#include <hip/hip_runtime.h>
#include <hip/hip_bf16.h>
#include <stdint.h>

// ---------------- problem constants ----------------
#define NN    10000      // N_NODES
#define NE    640000     // N_EDGES
#define DN    128        // D_NODE
#define DE    128        // D_EDGE
#define HE    256        // H_EDGE
#define DIN   384        // 2*DN + DE  (edge MLP input)  == 3*DN (node MLP input)

typedef __bf16 bf16;
typedef __attribute__((ext_vector_type(16))) __bf16 v16bf;
typedef __attribute__((ext_vector_type(8)))  float  v8f;
typedef __attribute__((ext_vector_type(4)))  unsigned int v4u;
typedef __attribute__((ext_vector_type(2)))  unsigned int v2u;

union V16 { v4u q[2]; v16bf v; };   // 32 bytes, maps to 8 VGPRs of packed bf16

static __device__ inline unsigned pk2(float a, float b) {
    unsigned short lo = __builtin_bit_cast(unsigned short, (bf16)a);
    unsigned short hi = __builtin_bit_cast(unsigned short, (bf16)b);
    return (unsigned)lo | ((unsigned)hi << 16);
}

// ---------------- prep kernels ----------------
__global__ void zero_f32(float* p, int n) {
    int i = blockIdx.x * blockDim.x + threadIdx.x;
    if (i < n) p[i] = 0.0f;
}

__global__ void cvt_bf16(const float* __restrict__ in, bf16* __restrict__ out, int n) {
    int i = blockIdx.x * blockDim.x + threadIdx.x;
    if (i < n) out[i] = (bf16)in[i];
}

// in: row-major [K][N] fp32 ; out: row-major [N][K] bf16  (B operand layout)
__global__ void cvt_transpose(const float* __restrict__ in, bf16* __restrict__ out,
                              int K, int N) {
    int i = blockIdx.x * blockDim.x + threadIdx.x;
    if (i < K * N) {
        int k = i / N, n = i - k * N;
        out[(size_t)n * K + k] = (bf16)in[i];
    }
}

// ---------------- WMMA helpers ----------------
// Per-lane A/B fragment loader following the CDNA5 16-bit operand layout:
// lanes 0-15  : K = kc*32 + {0..7}  and kc*32 + {16..23}
// lanes 16-31 : K = kc*32 + {8..15} and kc*32 + {24..31}
// A: lane row = lane&15, from LDS row-major [16][K].  B: lane col = lane&15,
// from global row-major [N][K] (pre-transposed weights).

static __device__ inline v16bf load_frag_lds(const bf16* base, int row, int rowStride,
                                             int k0, int k1) {
    V16 f;
    f.q[0] = *(const v4u*)(base + (size_t)row * rowStride + k0);
    f.q[1] = *(const v4u*)(base + (size_t)row * rowStride + k1);
    return f.v;
}

static __device__ inline v16bf load_frag_glb(const bf16* __restrict__ base, int row,
                                             int rowStride, int k0, int k1) {
    V16 f;
    f.q[0] = *(const v4u*)(base + (size_t)row * rowStride + k0);
    f.q[1] = *(const v4u*)(base + (size_t)row * rowStride + k1);
    return f.v;
}

// ---------------- edge-message kernel ----------------
// One wave per 16-edge tile; both directions share one gathered LDS tile.
// grid: 20000 blocks x 64 threads (2 waves) -> 40000 tiles == NE/16 exactly.
__global__ __launch_bounds__(64) void edge_msgs(
    const bf16* __restrict__ nodebf,       // [NN][DN] bf16
    const int*  __restrict__ from_idx,
    const int*  __restrict__ to_idx,
    const float* __restrict__ edge_feats,  // [NE][DE] fp32
    const bf16* __restrict__ mw1T, const float* __restrict__ mb1,
    const bf16* __restrict__ mw2T, const float* __restrict__ mb2,
    const bf16* __restrict__ rw1T, const float* __restrict__ rb1,
    const bf16* __restrict__ rw2T, const float* __restrict__ rb2,
    float* __restrict__ agg)               // [NN][HE] fp32, pre-zeroed
{
    __shared__ char smem[2 * (DIN * 16 * 2 + HE * 16 * 2) + 2 * 32 * 4];

    const int wave = threadIdx.x >> 5;
    const int lane = threadIdx.x & 31;
    const int m    = lane & 15;      // A row / B col / D col
    const int half = lane >> 4;

    char* my      = smem + wave * (DIN * 16 * 2 + HE * 16 * 2);
    bf16* einLds  = (bf16*)my;                         // [16][384]
    bf16* hLds    = (bf16*)(my + DIN * 16 * 2);        // [16][256]
    int*  idxLds  = (int*)(smem + 2 * (DIN * 16 * 2 + HE * 16 * 2)) + wave * 32;

    const int  tile  = blockIdx.x * 2 + wave;          // < 40000 always
    const long ebase = (long)tile * 16;

    // ---- stash indices (lanes 0-15: from, 16-31: to) ----
    idxLds[lane] = (lane < 16) ? from_idx[ebase + m] : to_idx[ebase + m];

    // ---- gather 16 edges: [node[from] | node[to] | edge_feat] as bf16 ----
    for (int e = 0; e < 16; ++e) {
        const int fi = idxLds[e];
        const int ti = idxLds[16 + e];
        // node rows: 128 bf16 = 32 lanes x 4 bf16 (8B) each
        *((v2u*)(einLds + e * DIN) + lane) =
            *((const v2u*)(nodebf + (size_t)fi * DN) + lane);
        *((v2u*)(einLds + e * DIN + DN) + lane) =
            *((const v2u*)(nodebf + (size_t)ti * DN) + lane);
        // edge row: fp32 -> bf16 on the fly
        float4 f = ((const float4*)(edge_feats + (ebase + e) * (size_t)DE))[lane];
        v2u p; p.x = pk2(f.x, f.y); p.y = pk2(f.z, f.w);
        *((v2u*)(einLds + e * DIN + 2 * DN) + lane) = p;
    }

    for (int dir = 0; dir < 2; ++dir) {
        const bf16*  W1 = dir ? rw1T : mw1T;
        const bf16*  W2 = dir ? rw2T : mw2T;
        const float* b1 = dir ? rb1  : mb1;
        const float* b2 = dir ? rb2  : mb2;

        // ---- layer 1: [16,384] @ W1 -> h [16,256], bias + ReLU, restage to LDS
        for (int nt = 0; nt < 16; ++nt) {
            v8f acc = {};
            const int n = nt * 16 + m;
            #pragma unroll
            for (int c = 0; c < 12; ++c) {
                int k0 = c * 32 + half * 8;
                int k1 = k0 + 16;
                // reverse direction swaps the two node segments of edge_in
                int ak0 = dir ? (k0 < 128 ? k0 + 128 : (k0 < 256 ? k0 - 128 : k0)) : k0;
                int ak1 = dir ? (k1 < 128 ? k1 + 128 : (k1 < 256 ? k1 - 128 : k1)) : k1;
                v16bf a = load_frag_lds(einLds, m, DIN, ak0, ak1);
                v16bf b = load_frag_glb(W1, n, DIN, k0, k1);
                acc = __builtin_amdgcn_wmma_f32_16x16x32_bf16(
                    false, a, false, b, (short)0, acc, false, false);
            }
            const float bias = b1[n];
            #pragma unroll
            for (int j = 0; j < 8; ++j) {
                float v = acc[j] + bias;
                v = v > 0.0f ? v : 0.0f;
                hLds[(size_t)(j + half * 8) * HE + n] = (bf16)v;   // D layout -> A layout
            }
        }

        // ---- layer 2: h [16,256] @ W2 -> msgs [16,256], bias, atomic scatter
        for (int nt = 0; nt < 16; ++nt) {
            v8f acc = {};
            const int n = nt * 16 + m;
            #pragma unroll
            for (int c = 0; c < 8; ++c) {
                int k0 = c * 32 + half * 8;
                int k1 = k0 + 16;
                v16bf a = load_frag_lds(hLds, m, HE, k0, k1);
                v16bf b = load_frag_glb(W2, n, HE, k0, k1);
                acc = __builtin_amdgcn_wmma_f32_16x16x32_bf16(
                    false, a, false, b, (short)0, acc, false, false);
            }
            const float bias = b2[n];
            #pragma unroll
            for (int j = 0; j < 8; ++j) {
                const int mm  = j + half * 8;                      // edge within tile
                const int dst = dir ? idxLds[mm] : idxLds[16 + mm];
                atomicAdd(agg + (size_t)dst * HE + n, acc[j] + bias);
            }
        }
    }
}

// ---------------- node-update kernel ----------------
// grid: 313 blocks x 64 threads (2 waves) -> 626 tiles, 625 used (NN/16).
__global__ __launch_bounds__(64) void node_update(
    const float* __restrict__ node_states,  // [NN][DN] fp32
    const float* __restrict__ agg,          // [NN][HE] fp32
    const bf16* __restrict__ nw1T, const float* __restrict__ nb1,
    const bf16* __restrict__ nw2T, const float* __restrict__ nb2,
    float* __restrict__ out)                // [NN][DN] fp32
{
    __shared__ char smem[2 * (DIN * 16 * 2 + HE * 16 * 2)];

    const int wave = threadIdx.x >> 5;
    const int lane = threadIdx.x & 31;
    const int m    = lane & 15;
    const int half = lane >> 4;
    const int tile = blockIdx.x * 2 + wave;
    if (tile >= NN / 16) return;

    char* my     = smem + wave * (DIN * 16 * 2 + HE * 16 * 2);
    bf16* xLds   = (bf16*)my;                    // [16][384] = [agg(256)|node(128)]
    bf16* hLds   = (bf16*)(my + DIN * 16 * 2);   // [16][256]
    const int nbase = tile * 16;

    // ---- gather x rows, fp32 -> bf16 ----
    for (int r = 0; r < 16; ++r) {
        const int node = nbase + r;
        #pragma unroll
        for (int t = 0; t < 3; ++t) {
            const int f4 = lane + t * 32;        // 96 float4 per row
            float4 v = (f4 < 64)
                ? ((const float4*)(agg + (size_t)node * HE))[f4]
                : ((const float4*)(node_states + (size_t)node * DN))[f4 - 64];
            v2u p; p.x = pk2(v.x, v.y); p.y = pk2(v.z, v.w);
            *((v2u*)(xLds + r * DIN) + f4) = p;
        }
    }

    // ---- layer 1: [16,384] @ nw1 -> h [16,256], bias + ReLU ----
    for (int nt = 0; nt < 16; ++nt) {
        v8f acc = {};
        const int n = nt * 16 + m;
        #pragma unroll
        for (int c = 0; c < 12; ++c) {
            int k0 = c * 32 + half * 8;
            int k1 = k0 + 16;
            v16bf a = load_frag_lds(xLds, m, DIN, k0, k1);
            v16bf b = load_frag_glb(nw1T, n, DIN, k0, k1);
            acc = __builtin_amdgcn_wmma_f32_16x16x32_bf16(
                false, a, false, b, (short)0, acc, false, false);
        }
        const float bias = nb1[n];
        #pragma unroll
        for (int j = 0; j < 8; ++j) {
            float v = acc[j] + bias;
            v = v > 0.0f ? v : 0.0f;
            hLds[(size_t)(j + half * 8) * HE + n] = (bf16)v;
        }
    }

    // ---- layer 2: [16,256] @ nw2 -> [16,128], residual add, store ----
    for (int nt = 0; nt < 8; ++nt) {
        v8f acc = {};
        const int col = nt * 16 + m;
        #pragma unroll
        for (int c = 0; c < 8; ++c) {
            int k0 = c * 32 + half * 8;
            int k1 = k0 + 16;
            v16bf a = load_frag_lds(hLds, m, HE, k0, k1);
            v16bf b = load_frag_glb(nw2T, col, HE, k0, k1);
            acc = __builtin_amdgcn_wmma_f32_16x16x32_bf16(
                false, a, false, b, (short)0, acc, false, false);
        }
        const float bias = nb2[col];
        #pragma unroll
        for (int j = 0; j < 8; ++j) {
            const int node = nbase + j + half * 8;
            out[(size_t)node * DN + col] =
                node_states[(size_t)node * DN + col] + acc[j] + bias;
        }
    }
}

// ---------------- host launch ----------------
extern "C" void kernel_launch(void* const* d_in, const int* in_sizes, int n_in,
                              void* d_out, int out_size, void* d_ws, size_t ws_size,
                              hipStream_t stream) {
    const float* node_states = (const float*)d_in[0];
    const int*   from_idx    = (const int*)  d_in[1];
    const int*   to_idx      = (const int*)  d_in[2];
    const float* edge_feats  = (const float*)d_in[3];
    const float* mw1 = (const float*)d_in[4];  const float* mb1 = (const float*)d_in[5];
    const float* mw2 = (const float*)d_in[6];  const float* mb2 = (const float*)d_in[7];
    const float* rw1 = (const float*)d_in[8];  const float* rb1 = (const float*)d_in[9];
    const float* rw2 = (const float*)d_in[10]; const float* rb2 = (const float*)d_in[11];
    const float* nw1 = (const float*)d_in[12]; const float* nb1 = (const float*)d_in[13];
    const float* nw2 = (const float*)d_in[14]; const float* nb2 = (const float*)d_in[15];
    float* out = (float*)d_out;

    // workspace layout (all offsets 256B-aligned)
    char*  ws     = (char*)d_ws;
    float* agg    = (float*)(ws);                      // NN*HE fp32   = 10,240,000 B
    bf16*  nodebf = (bf16*)(ws + 10240000);            // NN*DN bf16   =  2,560,000 B
    bf16*  mw1T   = (bf16*)(ws + 12800000);            // 256x384 bf16 =    196,608 B
    bf16*  mw2T   = (bf16*)(ws + 12996608);            // 256x256 bf16 =    131,072 B
    bf16*  rw1T   = (bf16*)(ws + 13127680);            // 196,608 B
    bf16*  rw2T   = (bf16*)(ws + 13324288);            // 131,072 B
    bf16*  nw1T   = (bf16*)(ws + 13455360);            // 196,608 B
    bf16*  nw2T   = (bf16*)(ws + 13651968);            // 128x256 bf16 = 65,536 B

    const int T = 256;
    zero_f32<<<(NN * HE + T - 1) / T, T, 0, stream>>>(agg, NN * HE);
    cvt_bf16<<<(NN * DN + T - 1) / T, T, 0, stream>>>(node_states, nodebf, NN * DN);
    cvt_transpose<<<(DIN * HE + T - 1) / T, T, 0, stream>>>(mw1, mw1T, DIN, HE);
    cvt_transpose<<<(HE * HE  + T - 1) / T, T, 0, stream>>>(mw2, mw2T, HE, HE);
    cvt_transpose<<<(DIN * HE + T - 1) / T, T, 0, stream>>>(rw1, rw1T, DIN, HE);
    cvt_transpose<<<(HE * HE  + T - 1) / T, T, 0, stream>>>(rw2, rw2T, HE, HE);
    cvt_transpose<<<(DIN * HE + T - 1) / T, T, 0, stream>>>(nw1, nw1T, DIN, HE);
    cvt_transpose<<<(HE * DN  + T - 1) / T, T, 0, stream>>>(nw2, nw2T, HE, DN);

    edge_msgs<<<NE / 16 / 2, 64, 0, stream>>>(
        nodebf, from_idx, to_idx, edge_feats,
        mw1T, mb1, mw2T, mb2, rw1T, rb1, rw2T, rb2, agg);

    node_update<<<(NN / 16 + 1) / 2, 64, 0, stream>>>(
        node_states, agg, nw1T, nb1, nw2T, nb2, out);
}
